// VQVAEres_40183714022180
// MI455X (gfx1250) — matrix operand order repaired and checked by reading, not compiled, not run on hardware
//
#include <hip/hip_runtime.h>
#include <stdint.h>

// VQ-VAE forward for MI455X (gfx1250, wave32, WMMA).
// GEMM-shaped convs run as implicit-GEMM on v_wmma_f32_16x16x32_bf16 with
// NHWC bf16 activations (b128 A loads) and [tap][CoutPad][Cin] bf16 weights
// (b128 B loads). Conv geometry is templated (power-of-two spatial dims) so
// all index math is shifts; each wave computes a 16x64 output strip (4 B tiles
// per A fragment) to amortize the scattered activation gathers.

#define HID    256
#define KCLUST 512
#define EPSBN  1e-5f

typedef __attribute__((ext_vector_type(16))) __bf16 v16bf;
typedef __attribute__((ext_vector_type(8)))  __bf16 v8bf;
typedef __attribute__((ext_vector_type(8)))  float  v8f;

static __device__ __forceinline__ __bf16 f2bf(float v) { return (__bf16)v; }

// ---------------------------------------------------------------------------
// Weight packing: conv (tr=0) src [O][I][kh][kw], convT (tr=1) src [I][O][kh][kw]
//   -> dst [t][P][I] bf16, zero-padded for co >= O.
// ---------------------------------------------------------------------------
__global__ __launch_bounds__(256) void pack_w_kernel(const float* __restrict__ src,
    __bf16* __restrict__ dst, int O, int I, int KT, int P, int tr)
{
    int idx = blockIdx.x * 256 + threadIdx.x;
    int total = KT * P * I;
    if (idx >= total) return;
    int ci = idx % I;
    int co = (idx / I) % P;
    int t  = idx / (I * P);
    float v = 0.f;
    if (co < O)
        v = tr ? src[((size_t)ci * O + co) * KT + t]
               : src[((size_t)co * I + ci) * KT + t];
    dst[((size_t)t * P + co) * I + ci] = f2bf(v);
}

// ---------------------------------------------------------------------------
// Implicit-GEMM conv / transposed-conv on WMMA bf16, templated geometry.
// in: NHWC bf16 [16,HIN,WIN,CIN]; wt: [KH*KW][COUTP][CIN] bf16;
// out: NCHW f32 [16,Cout,HOUT,WOUT].
// Each wave: one 16(M) x (16*NCO)(N) strip; 4 waves/block.
// A frag (16x32 bf16): M=lane&15, K = 16*(lane>>4) + e
// B frag (32x16 bf16): N=lane&15, K = 16*(lane>>4) + e
// D frag (16x16 f32) : N=lane&15, M = r + 8*(lane>>4)
// ---------------------------------------------------------------------------
template<int KH, int KW, int STRIDE, int PAD, int TR,
         int CIN, int HIN, int WIN, int HOUT, int WOUT, int COUTP, int NCO>
__global__ __launch_bounds__(128) void conv_wmma_kernel(
    const __bf16* __restrict__ in, const __bf16* __restrict__ wt,
    const float* __restrict__ bias, float* __restrict__ out, int Cout)
{
    constexpr int HWo = HOUT * WOUT;
    constexpr int KT  = KH * KW;

    const int lane = threadIdx.x & 31;
    const int wave = threadIdx.x >> 5;
    const int half = lane >> 4;
    const int l16  = lane & 15;
    const unsigned mbase  = (blockIdx.x * 4 + wave) * 16;
    const int      cobase = blockIdx.y * (16 * NCO);

    // A-row output pixel for this lane (all divides are power-of-two -> shifts)
    const unsigned m  = mbase + l16;
    const unsigned n  = m / (unsigned)HWo;
    const unsigned hw = m % (unsigned)HWo;
    const int oh = (int)(hw / (unsigned)WOUT);
    const int ow = (int)(hw % (unsigned)WOUT);

    const int kOffA = half * 16;   // A K base (16 contiguous)
    const int kOffB = half * 16;   // B K base (16 contiguous)

    const __bf16* wco[NCO];
    #pragma unroll
    for (int j = 0; j < NCO; ++j)
        wco[j] = wt + ((size_t)(cobase + j * 16 + l16)) * CIN + kOffB;

    v8f acc[NCO] = {};

    for (int t = 0; t < KT; ++t) {
        const int r = t / KW, s = t - r * KW;
        int ih, iw; bool valid;
        if (TR == 0) {
            ih = oh * STRIDE - PAD + r;
            iw = ow * STRIDE - PAD + s;
            valid = (ih >= 0) && (ih < HIN) && (iw >= 0) && (iw < WIN);
        } else {
            // conv_transpose stride-2: oh = 2*ih - PAD + r  ->  ih = (oh+PAD-r)/2
            const int th = oh + PAD - r, tw = ow + PAD - s;
            valid = (th >= 0) && (tw >= 0) && ((th & 1) == 0) && ((tw & 1) == 0);
            ih = th >> 1; iw = tw >> 1;
            valid = valid && (ih < HIN) && (iw < WIN);
        }
        const __bf16* abase = in + (((size_t)n * HIN + ih) * WIN + iw) * CIN + kOffA;
        const size_t  wTap  = (size_t)t * ((size_t)COUTP * CIN);

        #pragma unroll 4
        for (int kc = 0; kc < CIN; kc += 32) {
            v16bf a = {};
            if (valid) {
                const v8bf alo = *(const v8bf*)(abase + kc);
                const v8bf ahi = *(const v8bf*)(abase + kc + 8);
                #pragma unroll
                for (int e = 0; e < 8; ++e) { a[e] = alo[e]; a[e + 8] = ahi[e]; }
            }
            #pragma unroll
            for (int j = 0; j < NCO; ++j) {
                const v16bf b = *(const v16bf*)(wco[j] + wTap + kc);
                acc[j] = __builtin_amdgcn_wmma_f32_16x16x32_bf16(
                    false, a, false, b, (short)0, acc[j], false, false);
            }
        }
    }

    #pragma unroll
    for (int j = 0; j < NCO; ++j) {
        const int co = cobase + j * 16 + l16;
        if (co < Cout) {
            const float bv = bias ? bias[co] : 0.f;
            #pragma unroll
            for (int r8 = 0; r8 < 8; ++r8) {
                const unsigned mm  = mbase + half * 8 + r8;
                const unsigned n2  = mm / (unsigned)HWo;
                const unsigned hw2 = mm % (unsigned)HWo;
                out[((size_t)n2 * Cout + co) * HWo + hw2] = acc[j][r8] + bv;
            }
        }
    }
}

template<int KH, int KW, int STRIDE, int PAD, int TR,
         int CIN, int HIN, int WIN, int HOUT, int WOUT, int COUTP, int NCO>
static void launch_conv(const __bf16* in, const __bf16* wt, const float* bias,
                        float* out, int Cout, hipStream_t st)
{
    constexpr int M = 16 * HOUT * WOUT;     // batch 16; always multiple of 64
    dim3 grid(M / 64, COUTP / (16 * NCO));
    conv_wmma_kernel<KH, KW, STRIDE, PAD, TR, CIN, HIN, WIN, HOUT, WOUT, COUTP, NCO>
        <<<grid, 128, 0, st>>>(in, wt, bias, out, Cout);
}

// ---------------------------------------------------------------------------
// Direct f32 conv for encoder stage 1 (Cin=3, cheap: ~1.6 GFLOP).
// x [16,3,128,128] -> out [16,256,64,64], k=4 s=2 p=1.
// ---------------------------------------------------------------------------
__global__ __launch_bounds__(256) void enc_conv1_kernel(const float* __restrict__ x,
    const float* __restrict__ w, const float* __restrict__ b, float* __restrict__ out)
{
    const int idx = blockIdx.x * 256 + threadIdx.x;   // 2^24 total
    const int ow = idx & 63;
    const int oh = (idx >> 6) & 63;
    const int co = (idx >> 12) & 255;
    const int n  = idx >> 20;
    float acc = b[co];
    for (int ci = 0; ci < 3; ++ci)
        for (int r = 0; r < 4; ++r) {
            const int ih = oh * 2 - 1 + r;
            if (ih < 0 || ih >= 128) continue;
            for (int s = 0; s < 4; ++s) {
                const int iw = ow * 2 - 1 + s;
                if (iw < 0 || iw >= 128) continue;
                acc += x[(((size_t)n * 3 + ci) * 128 + ih) * 128 + iw]
                     * w[(((size_t)co * 3 + ci) * 4 + r) * 4 + s];
            }
        }
    out[idx] = acc;
}

// ---------------------------------------------------------------------------
// BatchNorm batch statistics (training mode): mean/var over (N,H,W) per channel.
// ---------------------------------------------------------------------------
__global__ __launch_bounds__(256) void bn_stats_kernel(const float* __restrict__ x,
    float* __restrict__ mean, float* __restrict__ var, int N, int C, int HW)
{
    __shared__ float ss[256], sq[256];
    const int c = blockIdx.x;
    const int total = N * HW;
    float s = 0.f, q = 0.f;
    for (int i = threadIdx.x; i < total; i += 256) {
        const int n = i / HW, j = i - n * HW;
        const float v = x[((size_t)n * C + c) * HW + j];
        s += v; q += v * v;
    }
    ss[threadIdx.x] = s; sq[threadIdx.x] = q;
    __syncthreads();
    for (int off = 128; off > 0; off >>= 1) {
        if (threadIdx.x < off) {
            ss[threadIdx.x] += ss[threadIdx.x + off];
            sq[threadIdx.x] += sq[threadIdx.x + off];
        }
        __syncthreads();
    }
    if (threadIdx.x == 0) {
        const float m = ss[0] / (float)total;
        mean[c] = m;
        var[c]  = sq[0] / (float)total - m * m;
    }
}

// BN apply (+ optional ReLU); writes f32 NCHW and/or bf16 NHWC (next GEMM input).
__global__ __launch_bounds__(256) void bn_apply_kernel(const float* __restrict__ x,
    const float* __restrict__ mean, const float* __restrict__ var,
    const float* __restrict__ g, const float* __restrict__ be,
    float* __restrict__ of32, __bf16* __restrict__ obf,
    int C, int HW, int total, int relu)
{
    const int idx = blockIdx.x * 256 + threadIdx.x;
    if (idx >= total) return;
    const int hw = idx % HW;
    const int c  = (idx / HW) % C;
    const int n  = idx / (HW * C);
    float y = (x[idx] - mean[c]) * rsqrtf(var[c] + EPSBN) * g[c] + be[c];
    if (relu) y = fmaxf(y, 0.f);
    if (of32) of32[idx] = y;
    if (obf)  obf[((size_t)n * HW + hw) * C + c] = f2bf(y);
}

// NCHW f32 -> NHWC bf16 (optional ReLU).
__global__ __launch_bounds__(256) void cvt_bf16_kernel(const float* __restrict__ x,
    __bf16* __restrict__ o, int C, int HW, int total, int relu)
{
    const int idx = blockIdx.x * 256 + threadIdx.x;
    if (idx >= total) return;
    const int hw = idx % HW;
    const int c  = (idx / HW) % C;
    const int n  = idx / (HW * C);
    float v = x[idx];
    if (relu) v = fmaxf(v, 0.f);
    o[((size_t)n * HW + hw) * C + c] = f2bf(v);
}

__global__ __launch_bounds__(256) void copy_f32_kernel(const float* __restrict__ a,
    float* __restrict__ o, int total)
{
    const int idx = blockIdx.x * 256 + threadIdx.x;
    if (idx < total) o[idx] = a[idx];
}

__global__ __launch_bounds__(256) void add_f32_kernel(const float* __restrict__ a,
    const float* __restrict__ b, float* __restrict__ o, int total)
{
    const int idx = blockIdx.x * 256 + threadIdx.x;
    if (idx < total) o[idx] = a[idx] + b[idx];
}

// ||E_j||^2 per codebook entry.
__global__ __launch_bounds__(256) void vq_ee_kernel(const float* __restrict__ emb,
    float* __restrict__ ee)
{
    const int j = blockIdx.x * 256 + threadIdx.x;
    if (j >= KCLUST) return;
    const float* e = emb + (size_t)j * HID;
    float s = 0.f;
    for (int c = 0; c < HID; ++c) s += e[c] * e[c];
    ee[j] = s;
}

// argmin_j (||E_j||^2 - 2 z.E_j)  (||z||^2 constant per point -> dropped);
// gathers z_q = E[idx] (NCHW f32) and writes int32 latents.
__global__ __launch_bounds__(256) void vq_argmin_kernel(const float* __restrict__ dot,
    const float* __restrict__ ee, const float* __restrict__ emb,
    float* __restrict__ zq, int* __restrict__ lat, int N, int HW)
{
    const int p = blockIdx.x * 256 + threadIdx.x;
    if (p >= N * HW) return;
    const int n = p / HW, j0 = p - n * HW;
    const float* dp = dot + (size_t)n * KCLUST * HW + j0;
    float best = 3.4e38f; int bi = 0;
    for (int j = 0; j < KCLUST; ++j) {
        const float v = ee[j] - 2.f * dp[(size_t)j * HW];
        if (v < best) { best = v; bi = j; }
    }
    lat[p] = bi;
    const float* ev = emb + (size_t)bi * HID;
    float* zp = zq + (size_t)n * HID * HW + j0;
    for (int c = 0; c < HID; ++c) zp[(size_t)c * HW] = ev[c];
}

// ---------------------------------------------------------------------------
// Host orchestration
// ---------------------------------------------------------------------------
extern "C" void kernel_launch(void* const* d_in, const int* in_sizes, int n_in,
                              void* d_out, int out_size, void* d_ws, size_t ws_size,
                              hipStream_t stream)
{
    (void)in_sizes; (void)n_in; (void)out_size; (void)ws_size;
    auto F = [&](int i) { return (const float*)d_in[i]; };

    // setup_inputs() insertion order: x, then params (res(): w1,b1,g1,be1,w2,b2,g2,be2)
    const float* x       = F(0);
    const float* enc_w1  = F(1);
    const float* enc_b1  = F(2);
    const float* enc_g1  = F(3);
    const float* enc_be1 = F(4);
    const float* enc_w2  = F(5);
    const float* enc_b2  = F(6);
    const int er1 = 7, er2 = 15;
    const float* emb     = F(23);
    const int dr1 = 24, dr2 = 32;
    const float* dec_w1  = F(40);
    const float* dec_b1  = F(41);
    const float* dec_g1  = F(42);
    const float* dec_be1 = F(43);
    const float* dec_w2  = F(44);
    const float* dec_b2  = F(45);

    // workspace carve (bump allocator, 256B aligned)
    char* p = (char*)d_ws;
    auto alloc = [&](size_t bytes) -> void* {
        void* r = (void*)p; p += (bytes + 255) & ~(size_t)255; return r;
    };
    const size_t E64 = (size_t)16 * 256 * 64 * 64;   // 16,777,216
    const size_t E32 = (size_t)16 * 256 * 32 * 32;   //  4,194,304
    float*  A    = (float*) alloc(E64 * 4);          // 64x64 f32 stage
    __bf16* Abf  = (__bf16*)alloc(E64 * 2);          // 64x64 bf16 NHWC stage
    float*  B1   = (float*) alloc(E32 * 4);
    float*  B2   = (float*) alloc(E32 * 4);
    float*  B3   = (float*) alloc(E32 * 4);
    __bf16* Bbf  = (__bf16*)alloc(E32 * 2);
    float*  dotb = (float*) alloc((size_t)16 * KCLUST * 1024 * 4);
    __bf16* wEnc2 = (__bf16*)alloc((size_t)16 * 256 * 256 * 2);
    __bf16* wRes[4][2];
    for (int i = 0; i < 4; ++i) {
        wRes[i][0] = (__bf16*)alloc((size_t)9 * 256 * 256 * 2);
        wRes[i][1] = (__bf16*)alloc((size_t)1 * 256 * 256 * 2);
    }
    __bf16* wDec1 = (__bf16*)alloc((size_t)16 * 256 * 256 * 2);
    __bf16* wDec2 = (__bf16*)alloc((size_t)16 * 16  * 256 * 2);
    __bf16* wEmb  = (__bf16*)alloc((size_t)1  * 512 * 256 * 2);
    float* meanb = (float*)alloc(256 * 4);
    float* varb  = (float*)alloc(256 * 4);
    float* eeb   = (float*)alloc(KCLUST * 4);

    // output slices: (x_tilde, z_e, z_q, latents)
    float* xt   = (float*)d_out;
    float* ze_o = xt + (size_t)16 * 3 * 128 * 128;
    float* zq_o = ze_o + E32;
    int*   lat  = (int*)(zq_o + E32);

    // ---- pack weights to [tap][CoutPad][Cin] bf16 ----
    auto pack = [&](const float* src, __bf16* dst, int O, int I, int KT, int P, int tr) {
        const int total = KT * P * I;
        pack_w_kernel<<<(total + 255) / 256, 256, 0, stream>>>(src, dst, O, I, KT, P, tr);
    };
    pack(enc_w2, wEnc2, 256, 256, 16, 256, 0);
    const int resBase[4] = { er1, er2, dr1, dr2 };
    for (int i = 0; i < 4; ++i) {
        pack(F(resBase[i] + 0), wRes[i][0], 256, 256, 9, 256, 0);   // w1 3x3
        pack(F(resBase[i] + 4), wRes[i][1], 256, 256, 1, 256, 0);   // w2 1x1
    }
    pack(emb,    wEmb,  512, 256, 1,  512, 0);
    pack(dec_w1, wDec1, 256, 256, 16, 256, 1);
    pack(dec_w2, wDec2,   3, 256, 16,  16, 1);

    const int T32 = (int)E32, T64 = (int)E64;

    // ---- encoder ----
    enc_conv1_kernel<<<65536, 256, 0, stream>>>(x, enc_w1, enc_b1, A);
    bn_stats_kernel<<<256, 256, 0, stream>>>(A, meanb, varb, 16, 256, 4096);
    bn_apply_kernel<<<T64 / 256, 256, 0, stream>>>(A, meanb, varb, enc_g1, enc_be1,
                                                   nullptr, Abf, 256, 4096, T64, 1);
    // enc conv2: 4x4 s2 p1, 64x64 -> 32x32
    launch_conv<4,4,2,1,0, 256,64,64,32,32, 256,4>(Abf, wEnc2, enc_b2, B1, 256, stream);

    auto resblock = [&](const float* xin, float* xout, int base, __bf16* w1t, __bf16* w2t) {
        const float* b1  = F(base + 1); const float* g1 = F(base + 2);
        const float* be1 = F(base + 3);
        const float* b2  = F(base + 5); const float* g2 = F(base + 6);
        const float* be2 = F(base + 7);
        cvt_bf16_kernel<<<T32 / 256, 256, 0, stream>>>(xin, Bbf, 256, 1024, T32, 1);
        launch_conv<3,3,1,1,0, 256,32,32,32,32, 256,4>(Bbf, w1t, b1, B2, 256, stream);
        bn_stats_kernel<<<256, 256, 0, stream>>>(B2, meanb, varb, 16, 256, 1024);
        bn_apply_kernel<<<T32 / 256, 256, 0, stream>>>(B2, meanb, varb, g1, be1,
                                                       nullptr, Bbf, 256, 1024, T32, 1);
        launch_conv<1,1,1,0,0, 256,32,32,32,32, 256,4>(Bbf, w2t, b2, B3, 256, stream);
        bn_stats_kernel<<<256, 256, 0, stream>>>(B3, meanb, varb, 16, 256, 1024);
        bn_apply_kernel<<<T32 / 256, 256, 0, stream>>>(B3, meanb, varb, g2, be2,
                                                       B3, nullptr, 256, 1024, T32, 0);
        add_f32_kernel<<<T32 / 256, 256, 0, stream>>>(xin, B3, xout, T32);
    };

    resblock(B1, B1, er1, wRes[0][0], wRes[0][1]);
    resblock(B1, B1, er2, wRes[1][0], wRes[1][1]);   // B1 = z_e

    // ---- VQ ----
    copy_f32_kernel<<<T32 / 256, 256, 0, stream>>>(B1, ze_o, T32);
    cvt_bf16_kernel<<<T32 / 256, 256, 0, stream>>>(B1, Bbf, 256, 1024, T32, 0);
    vq_ee_kernel<<<2, 256, 0, stream>>>(emb, eeb);
    // z_e . E^T as 1x1 conv with Cout=512
    launch_conv<1,1,1,0,0, 256,32,32,32,32, 512,4>(Bbf, wEmb, nullptr, dotb, 512, stream);
    vq_argmin_kernel<<<64, 256, 0, stream>>>(dotb, eeb, emb, zq_o, lat, 16, 1024);

    // ---- decoder (straight-through forward value = z_q) ----
    resblock(zq_o, B1, dr1, wRes[2][0], wRes[2][1]);
    resblock(B1,  B1, dr2, wRes[3][0], wRes[3][1]);
    cvt_bf16_kernel<<<T32 / 256, 256, 0, stream>>>(B1, Bbf, 256, 1024, T32, 0);
    // dec conv_transpose1: 32x32 -> 64x64
    launch_conv<4,4,2,1,1, 256,32,32,64,64, 256,4>(Bbf, wDec1, dec_b1, A, 256, stream);
    bn_stats_kernel<<<256, 256, 0, stream>>>(A, meanb, varb, 16, 256, 4096);
    bn_apply_kernel<<<T64 / 256, 256, 0, stream>>>(A, meanb, varb, dec_g1, dec_be1,
                                                   nullptr, Abf, 256, 4096, T64, 1);
    // dec conv_transpose2: 64x64 -> 128x128, Cout=3 (padded to 16)
    launch_conv<4,4,2,1,1, 256,64,64,128,128, 16,1>(Abf, wDec2, dec_b2, xt, 3, stream);
}
